// LocalFeatureFusion_1520418423083
// MI455X (gfx1250) — compile-verified, matrix operand order checked by reading, not run on hardware
//
#include <hip/hip_runtime.h>
#include <hip/hip_bf16.h>
#include <math.h>

#define Bc   8
#define NQ   2048
#define LK   2048
#define CH   256
#define KN   16
#define NLAY 2
#define FFm  4
#define RAD2 (0.12f * 0.12f)

typedef __attribute__((ext_vector_type(16))) __bf16          v16bf;
typedef __attribute__((ext_vector_type(8)))  float           v8f;
typedef __attribute__((ext_vector_type(8)))  unsigned short  v8u;
typedef __attribute__((ext_vector_type(16))) unsigned short  v16u;

union FragU { v16u u; v16bf b; };

__device__ __forceinline__ unsigned short f2bf(float x) {
  unsigned int u = __float_as_uint(x);
  u += 0x7fffu + ((u >> 16) & 1u);       // round-to-nearest-even
  return (unsigned short)(u >> 16);
}

// Load one 16-bit A/B WMMA fragment (16x32 / 32x16) for this lane.
// Per CDNA5 ISA 7.12.2: lane (hi = lane/16) holds K = base+hi*8 .. +7 in halves 0..7
// and K = base+16+hi*8 .. +7 in halves 8..15, for row/col = lane%16.
__device__ __forceinline__ v16bf load_frag(const unsigned short* p) {
  v8u lo = *(const v8u*)(p);
  v8u hi = *(const v8u*)(p + 16);
  FragU f;
  f.u = __builtin_shufflevector(lo, hi, 0,1,2,3,4,5,6,7,8,9,10,11,12,13,14,15);
  return f.b;
}

// ---------------------------------------------------------------------------
// bf16 WMMA GEMM: C[M,N] = A[M,K] * Bt[N,K]^T   (Bt stored row-major = B^T)
// Block = 256 threads = 8 waves. Wave w computes rows [blockRow + 16w, +16),
// cols [colBase, colBase+64) as four 16x16 WMMA tiles sharing the A fragment.
// MODE 0: f32 store, no bias.  MODE 1: +bias, exact GELU, bf16 store.
// ---------------------------------------------------------------------------
template <int MODE>
__global__ __launch_bounds__(256)
void gemm_bf16_kernel(const unsigned short* __restrict__ A,
                      const unsigned short* __restrict__ Bt,
                      void* __restrict__ Cout,
                      const float* __restrict__ bias,
                      int M, int N, int Kd) {
  const int lane = threadIdx.x & 31;
  const int wave = threadIdx.x >> 5;
  const int hi   = lane >> 4;
  const int lm   = lane & 15;
  const int rowTile = blockIdx.y * 128 + wave * 16;
  const int colBase = blockIdx.x * 64;

  const unsigned short* arow = A + (size_t)(rowTile + lm) * Kd + hi * 8;
  const unsigned short* brow0 = Bt + (size_t)(colBase +  0 + lm) * Kd + hi * 8;
  const unsigned short* brow1 = Bt + (size_t)(colBase + 16 + lm) * Kd + hi * 8;
  const unsigned short* brow2 = Bt + (size_t)(colBase + 32 + lm) * Kd + hi * 8;
  const unsigned short* brow3 = Bt + (size_t)(colBase + 48 + lm) * Kd + hi * 8;

  v8f acc0 = {}, acc1 = {}, acc2 = {}, acc3 = {};

  for (int k = 0; k < Kd; k += 32) {
    v16bf a  = load_frag(arow + k);
    v16bf b0 = load_frag(brow0 + k);
    v16bf b1 = load_frag(brow1 + k);
    v16bf b2 = load_frag(brow2 + k);
    v16bf b3 = load_frag(brow3 + k);
    acc0 = __builtin_amdgcn_wmma_f32_16x16x32_bf16(false, a, false, b0, (short)0, acc0, false, false);
    acc1 = __builtin_amdgcn_wmma_f32_16x16x32_bf16(false, a, false, b1, (short)0, acc1, false, false);
    acc2 = __builtin_amdgcn_wmma_f32_16x16x32_bf16(false, a, false, b2, (short)0, acc2, false, false);
    acc3 = __builtin_amdgcn_wmma_f32_16x16x32_bf16(false, a, false, b3, (short)0, acc3, false, false);
  }

  v8f accs[4] = {acc0, acc1, acc2, acc3};
#pragma unroll
  for (int s = 0; s < 4; ++s) {
    const int col = colBase + s * 16 + lm;
#pragma unroll
    for (int r = 0; r < 8; ++r) {
      const int m = rowTile + hi * 8 + r;   // C/D layout: VGPR r -> M = r + 8*hi
      float val = accs[s][r];
      if (MODE == 1) {
        val += bias[col];
        val = 0.5f * val * (1.0f + erff(val * 0.70710678118654752f));
        ((unsigned short*)Cout)[(size_t)m * N + col] = f2bf(val);
      } else {
        ((float*)Cout)[(size_t)m * N + col] = val;
      }
    }
  }
}

// ---------------------------------------------------------------------------
// KNN: per-query top-16 smallest d2 over LK points, LDS-tiled.
// ---------------------------------------------------------------------------
__global__ __launch_bounds__(256)
void knn_kernel(const float* __restrict__ q_xyz, const float* __restrict__ kv_xyz,
                int* __restrict__ idx, float* __restrict__ validf) {
  __shared__ float sx[256], sy[256], sz[256];
  const int b      = blockIdx.x / (NQ / 256);
  const int n      = (blockIdx.x % (NQ / 256)) * 256 + threadIdx.x;
  const size_t qo  = ((size_t)b * NQ + n) * 3;
  const float qx = q_xyz[qo], qy = q_xyz[qo + 1], qz = q_xyz[qo + 2];

  float dist[KN]; int ind[KN];
#pragma unroll
  for (int k = 0; k < KN; ++k) { dist[k] = 1e30f; ind[k] = 0; }

  for (int t = 0; t < LK; t += 256) {
    __syncthreads();
    const size_t ko = ((size_t)b * LK + t + threadIdx.x) * 3;
    sx[threadIdx.x] = kv_xyz[ko];
    sy[threadIdx.x] = kv_xyz[ko + 1];
    sz[threadIdx.x] = kv_xyz[ko + 2];
    __syncthreads();
    for (int jj = 0; jj < 256; ++jj) {
      const float dx = qx - sx[jj], dy = qy - sy[jj], dz = qz - sz[jj];
      const float d2 = dx * dx + dy * dy + dz * dz;
      if (d2 < dist[KN - 1]) {
        dist[KN - 1] = d2; ind[KN - 1] = t + jj;
#pragma unroll
        for (int s = KN - 1; s > 0; --s) {
          if (dist[s] < dist[s - 1]) {
            float td = dist[s]; dist[s] = dist[s - 1]; dist[s - 1] = td;
            int   ti = ind[s];  ind[s]  = ind[s - 1];  ind[s - 1]  = ti;
          }
        }
      }
    }
  }
  const size_t base = ((size_t)b * NQ + n) * KN;
#pragma unroll
  for (int k = 0; k < KN; ++k) {
    idx[base + k]    = ind[k];
    validf[base + k] = (dist[k] <= RAD2) ? 1.0f : 0.0f;
  }
}

// ---------------------------------------------------------------------------
// Weight prep
// ---------------------------------------------------------------------------
// WsaT[l][n][k] = sum_c W_src[l][k][c]*W_attn[l][c][n] (bf16, transposed for GEMM B)
__global__ void compose_wT_kernel(const float* __restrict__ W_src,
                                  const float* __restrict__ W_dst,
                                  const float* __restrict__ W_attn,
                                  unsigned short* __restrict__ WsaT,
                                  unsigned short* __restrict__ WdaT) {
  const int n = blockIdx.x * 16 + threadIdx.x;
  const int k = blockIdx.y * 16 + threadIdx.y;
  const int layer = blockIdx.z >> 1, which = blockIdx.z & 1;
  const float* Wx = (which ? W_dst : W_src) + (size_t)layer * CH * CH;
  const float* Wa = W_attn + (size_t)layer * CH * CH;
  float acc = 0.0f;
  for (int c = 0; c < CH; ++c) acc += Wx[(size_t)k * CH + c] * Wa[(size_t)c * CH + n];
  unsigned short* dst = (which ? WdaT : WsaT) + (size_t)layer * CH * CH;
  dst[(size_t)n * CH + k] = f2bf(acc);
}

// dst[n][k] = bf16(src[k][n]); src is [R][Ncols]
__global__ void transpose_bf16_kernel(const float* __restrict__ src,
                                      unsigned short* __restrict__ dst,
                                      int R, int Ncols) {
  const int i = blockIdx.x * 256 + threadIdx.x;
  if (i >= R * Ncols) return;
  const int n = i / R, k = i % R;
  dst[(size_t)n * R + k] = f2bf(src[(size_t)k * Ncols + n]);
}

__global__ void f32_to_bf16_kernel(const float* __restrict__ s,
                                   unsigned short* __restrict__ d, int cnt) {
  const int i = blockIdx.x * 256 + threadIdx.x;
  if (i < cnt) d[i] = f2bf(s[i]);
}

__global__ void init_out_kernel(const float* __restrict__ q, float* __restrict__ of,
                                unsigned short* __restrict__ ob, int cnt) {
  const int i = blockIdx.x * 256 + threadIdx.x;
  if (i < cnt) { const float v = q[i]; of[i] = v; ob[i] = f2bf(v); }
}

// ---------------------------------------------------------------------------
// Fused LN helper: block-wide mean/var over CH channels (blockDim == CH)
// ---------------------------------------------------------------------------
__device__ __forceinline__ float2 block_meanvar(float x, float* red, int c) {
  red[c] = x; __syncthreads();
  for (int s = CH / 2; s > 0; s >>= 1) { if (c < s) red[c] += red[c + s]; __syncthreads(); }
  const float mu = red[0] * (1.0f / CH); __syncthreads();
  const float d = x - mu;
  red[c] = d * d; __syncthreads();
  for (int s = CH / 2; s > 0; s >>= 1) { if (c < s) red[c] += red[c + s]; __syncthreads(); }
  const float var = red[0] * (1.0f / CH); __syncthreads();
  return make_float2(mu, var);
}

// ---------------------------------------------------------------------------
// Attention (per-channel softmax over K neighbors) + residual + LayerNorm1.
// alpha = relu(t_dst - t_src[j] + b_attn); masked softmax; conv = sum attn*v.
// One block per query row; thread = channel.
// ---------------------------------------------------------------------------
__global__ __launch_bounds__(CH)
void attn_ln_kernel(const float* __restrict__ t_dst, const float* __restrict__ t_src,
                    const float* __restrict__ vmat, const int* __restrict__ idx,
                    const float* __restrict__ validf, const float* __restrict__ b_attn,
                    const float* __restrict__ g, const float* __restrict__ be,
                    float* __restrict__ out_f32, unsigned short* __restrict__ out_bf16) {
  __shared__ float red[CH];
  const int row = blockIdx.x;            // b*NQ + n
  const int b   = row / NQ;
  const int c   = threadIdx.x;
  const float td = t_dst[(size_t)row * CH + c] + b_attn[c];

  float alpha[KN], vm[KN]; int jj[KN];
#pragma unroll
  for (int k = 0; k < KN; ++k) {
    const int j = idx[(size_t)row * KN + k];
    jj[k] = j;
    vm[k] = validf[(size_t)row * KN + k];
    float a = fmaxf(td - t_src[((size_t)b * LK + j) * CH + c], 0.0f);
    alpha[k] = (vm[k] > 0.0f) ? a : -1e9f;
  }
  float mx = -1e30f;
#pragma unroll
  for (int k = 0; k < KN; ++k) mx = fmaxf(mx, alpha[k]);
  float w[KN], ssum = 0.0f;
#pragma unroll
  for (int k = 0; k < KN; ++k) { w[k] = __expf(alpha[k] - mx); ssum += w[k]; }
  const float inv = 1.0f / ssum;
  float conv = 0.0f;
#pragma unroll
  for (int k = 0; k < KN; ++k)
    conv += (w[k] * inv * vm[k]) * vmat[((size_t)b * LK + jj[k]) * CH + c];

  const float x = out_f32[(size_t)row * CH + c] + conv;
  const float2 mv = block_meanvar(x, red, c);
  const float y = (x - mv.x) * rsqrtf(mv.y + 1e-5f) * g[c] + be[c];
  out_f32[(size_t)row * CH + c]  = y;
  out_bf16[(size_t)row * CH + c] = f2bf(y);
}

// out = LN(out + y + b2) ; writes f32 + bf16
__global__ __launch_bounds__(CH)
void ffn_ln_kernel(const float* __restrict__ ymat, const float* __restrict__ b2,
                   const float* __restrict__ g, const float* __restrict__ be,
                   float* __restrict__ out_f32, unsigned short* __restrict__ out_bf16) {
  __shared__ float red[CH];
  const int row = blockIdx.x;
  const int c   = threadIdx.x;
  const float x = out_f32[(size_t)row * CH + c] + ymat[(size_t)row * CH + c] + b2[c];
  const float2 mv = block_meanvar(x, red, c);
  const float y = (x - mv.x) * rsqrtf(mv.y + 1e-5f) * g[c] + be[c];
  out_f32[(size_t)row * CH + c]  = y;
  out_bf16[(size_t)row * CH + c] = f2bf(y);
}

// ---------------------------------------------------------------------------
extern "C" void kernel_launch(void* const* d_in, const int* in_sizes, int n_in,
                              void* d_out, int out_size, void* d_ws, size_t ws_size,
                              hipStream_t stream) {
  (void)in_sizes; (void)n_in; (void)out_size; (void)ws_size;
  const float* q_xyz   = (const float*)d_in[0];
  const float* q_feat  = (const float*)d_in[1];
  const float* kv_xyz  = (const float*)d_in[2];
  const float* kv_feat = (const float*)d_in[3];
  const float* W_src   = (const float*)d_in[4];
  const float* W_dst   = (const float*)d_in[5];
  const float* W_lin   = (const float*)d_in[6];
  const float* W_attn  = (const float*)d_in[7];
  const float* b_attn  = (const float*)d_in[8];
  const float* W1      = (const float*)d_in[9];
  const float* b1      = (const float*)d_in[10];
  const float* W2      = (const float*)d_in[11];
  const float* b2      = (const float*)d_in[12];
  const float* ln1_g   = (const float*)d_in[13];
  const float* ln1_b   = (const float*)d_in[14];
  const float* ln2_g   = (const float*)d_in[15];
  const float* ln2_b   = (const float*)d_in[16];

  float* out_f32 = (float*)d_out;              // [B,N,C] fp32, evolves in place

  // ---- workspace carve-up (256B aligned regions) ----
  char* w = (char*)d_ws;
  size_t off = 0;
  auto carve = [&](size_t bytes) -> char* {
    char* p = w + off;
    off += (bytes + 255) & ~(size_t)255;
    return p;
  };
  int*            idx      = (int*)carve((size_t)Bc * NQ * KN * 4);
  float*          validf   = (float*)carve((size_t)Bc * NQ * KN * 4);
  unsigned short* kv_bf16  = (unsigned short*)carve((size_t)Bc * LK * CH * 2);
  unsigned short* out_bf16 = (unsigned short*)carve((size_t)Bc * NQ * CH * 2);
  float*          t_src    = (float*)carve((size_t)Bc * LK * CH * 4);
  float*          t_dst    = (float*)carve((size_t)Bc * NQ * CH * 4);
  float*          vmat     = (float*)carve((size_t)Bc * LK * CH * 4);
  unsigned short* h_bf16   = (unsigned short*)carve((size_t)Bc * NQ * CH * FFm * 2);
  float*          ymat     = (float*)carve((size_t)Bc * NQ * CH * 4);
  unsigned short* WsaT     = (unsigned short*)carve((size_t)NLAY * CH * CH * 2);
  unsigned short* WdaT     = (unsigned short*)carve((size_t)NLAY * CH * CH * 2);
  unsigned short* WlinT    = (unsigned short*)carve((size_t)NLAY * CH * CH * 2);
  unsigned short* W1T      = (unsigned short*)carve((size_t)NLAY * CH * CH * FFm * 2);
  unsigned short* W2T      = (unsigned short*)carve((size_t)NLAY * CH * CH * FFm * 2);

  const int totQ = Bc * NQ * CH;      // 4,194,304
  const int totK = Bc * LK * CH;

  // ---- prep ----
  knn_kernel<<<Bc * (NQ / 256), 256, 0, stream>>>(q_xyz, kv_xyz, idx, validf);
  f32_to_bf16_kernel<<<(totK + 255) / 256, 256, 0, stream>>>(kv_feat, kv_bf16, totK);
  init_out_kernel<<<(totQ + 255) / 256, 256, 0, stream>>>(q_feat, out_f32, out_bf16, totQ);
  compose_wT_kernel<<<dim3(CH / 16, CH / 16, NLAY * 2), dim3(16, 16), 0, stream>>>(
      W_src, W_dst, W_attn, WsaT, WdaT);
  for (int i = 0; i < NLAY; ++i) {
    transpose_bf16_kernel<<<(CH * CH + 255) / 256, 256, 0, stream>>>(
        W_lin + (size_t)i * CH * CH, WlinT + (size_t)i * CH * CH, CH, CH);
    transpose_bf16_kernel<<<(CH * CH * FFm + 255) / 256, 256, 0, stream>>>(
        W1 + (size_t)i * CH * CH * FFm, W1T + (size_t)i * CH * CH * FFm, CH, CH * FFm);
    transpose_bf16_kernel<<<(CH * CH * FFm + 255) / 256, 256, 0, stream>>>(
        W2 + (size_t)i * CH * CH * FFm, W2T + (size_t)i * CH * CH * FFm, CH * FFm, CH);
  }

  const int Mrows = Bc * NQ;          // == Bc*LK == 16384
  const dim3 gemmBlk(256);
  const dim3 gridC(CH / 64, Mrows / 128);        // N=256
  const dim3 gridF(CH * FFm / 64, Mrows / 128);  // N=1024

  for (int i = 0; i < NLAY; ++i) {
    // t_src = kv @ (W_src W_attn); v = kv @ W_lin; t_dst = out @ (W_dst W_attn)
    gemm_bf16_kernel<0><<<gridC, gemmBlk, 0, stream>>>(
        kv_bf16, WsaT + (size_t)i * CH * CH, t_src, nullptr, Mrows, CH, CH);
    gemm_bf16_kernel<0><<<gridC, gemmBlk, 0, stream>>>(
        kv_bf16, WlinT + (size_t)i * CH * CH, vmat, nullptr, Mrows, CH, CH);
    gemm_bf16_kernel<0><<<gridC, gemmBlk, 0, stream>>>(
        out_bf16, WdaT + (size_t)i * CH * CH, t_dst, nullptr, Mrows, CH, CH);

    attn_ln_kernel<<<Bc * NQ, CH, 0, stream>>>(
        t_dst, t_src, vmat, idx, validf, b_attn + (size_t)i * CH,
        ln1_g + (size_t)i * CH, ln1_b + (size_t)i * CH, out_f32, out_bf16);

    // h = gelu(out @ W1 + b1)  (bf16 out);  y = h @ W2
    gemm_bf16_kernel<1><<<gridF, gemmBlk, 0, stream>>>(
        out_bf16, W1T + (size_t)i * CH * CH * FFm, h_bf16,
        b1 + (size_t)i * CH * FFm, Mrows, CH * FFm, CH);
    gemm_bf16_kernel<0><<<gridC, gemmBlk, 0, stream>>>(
        h_bf16, W2T + (size_t)i * CH * CH * FFm, ymat, nullptr, Mrows, CH, CH * FFm);

    ffn_ln_kernel<<<Bc * NQ, CH, 0, stream>>>(
        ymat, b2 + (size_t)i * CH, ln2_g + (size_t)i * CH, ln2_b + (size_t)i * CH,
        out_f32, out_bf16);
  }
}
